// encoder_11854109737685
// MI455X (gfx1250) — matrix operand order repaired and checked by reading, not compiled, not run on hardware
//
#include <hip/hip_runtime.h>
#include <hip/hip_bf16.h>

#define Bsz 256
#define TM1 512
#define Nin 256
#define Hh  256
#define NT  64            // N-tiles of 16 across gates width (4H = 1024)
#define KT  8             // K-tiles of 32 across 256
#define WAP 264           // padded row stride (bf16 halves) for wA/hA
#define GLP 260           // padded row stride (floats) for local gates

typedef __attribute__((ext_vector_type(16))) __bf16 v16bf;
typedef __attribute__((ext_vector_type(8)))  __bf16 v8bf;
typedef __attribute__((ext_vector_type(8)))  float  v8f;

static __device__ __forceinline__ v16bf cat16(v8bf lo, v8bf hi) {
  return __builtin_shufflevector(lo, hi, 0,1,2,3,4,5,6,7,8,9,10,11,12,13,14,15);
}
static __device__ __forceinline__ float sigf(float x) {
  return 1.0f / (1.0f + __expf(-x));
}
static __device__ __forceinline__ float tanhfast(float x) {
  return 2.0f / (1.0f + __expf(-2.0f * x)) - 1.0f;
}

// ---------------------------------------------------------------------------
// Kernel 1: attn[b,n] = softmax_n( sum_t input[b,t,n] * w_x[t] )
// (reference score's h/c terms are constant over n -> softmax drops them)
// ---------------------------------------------------------------------------
__global__ __launch_bounds__(256)
void attn_softmax_kernel(const float* __restrict__ input,
                         const float* __restrict__ attn_w,
                         float* __restrict__ attn_out) {
  const int b = blockIdx.x;
  const int n = threadIdx.x;                  // 256 threads
  const float* wx = attn_w + 2 * Hh;          // w_x = attn_w[2H : 2H+TM1]
  const float* base = input + (size_t)b * TM1 * Nin + n;
  float s = 0.0f;
  for (int t = 0; t < TM1; ++t) s += base[(size_t)t * Nin] * wx[t];

  __shared__ float red[256];
  red[n] = s; __syncthreads();
  for (int off = 128; off > 0; off >>= 1) {
    if (n < off) red[n] = fmaxf(red[n], red[n + off]);
    __syncthreads();
  }
  float mx = red[0]; __syncthreads();
  float e = __expf(s - mx);
  red[n] = e; __syncthreads();
  for (int off = 128; off > 0; off >>= 1) {
    if (n < off) red[n] += red[n + off];
    __syncthreads();
  }
  attn_out[b * Nin + n] = e / red[0];
}

// ---------------------------------------------------------------------------
// Kernel 2: pack W_ih / W_hh into per-lane WMMA B-fragments (bf16).
// Fragment (nt,kt) = 512 bf16 = 1KB at (nt*8+kt)*512:
//   dst[... + lane*16 + i] = W[n*256 + k],
//   n = nt*16 + (lane&15);  k = kt*32 + 8*(lane>>4) + (i<8 ? i : 8+i)
// Also zeroes the inter-WG barrier counters.
// ---------------------------------------------------------------------------
__global__ __launch_bounds__(512)
void pack_weights_kernel(const float* __restrict__ W_ih,
                         const float* __restrict__ W_hh,
                         __bf16* __restrict__ wsIH,
                         __bf16* __restrict__ wsHH,
                         unsigned* __restrict__ bar) {
  const int frag = blockIdx.x;          // 0..511 = nt*8 + kt
  const int mat  = blockIdx.y;          // 0 = ih, 1 = hh
  const int tid  = threadIdx.x;         // 0..511
  if (frag == 0 && mat == 0 && tid < 256) bar[tid] = 0u;
  const int l = tid >> 4, i = tid & 15;
  const int nt = frag >> 3, kt = frag & 7;
  const int n = nt * 16 + (l & 15);
  const int k = kt * 32 + 8 * (l >> 4) + ((i < 8) ? i : 8 + i);
  const float* W = mat ? W_hh : W_ih;
  __bf16* dst = mat ? wsHH : wsIH;
  dst[frag * 512 + tid] = (__bf16)W[n * 256 + k];
}

// ---------------------------------------------------------------------------
// Kernel 3: persistent recurrent kernel, 64 WGs = 16 groups x 4 partners.
// Group g owns batch rows [g*16, +16).  Partner s owns N-tiles nt == s (mod 4)
// -> it owns complete i/f/g/o gates for hidden units j with (j>>4)%4 == s,
//    so c and the LSTM cell update are WG-local; only h (4KB) is exchanged
//    through a double-buffered L2 buffer guarded by a 4-WG atomic barrier.
// Weights (2 x 128KB bf16 fragment slices) are LDS-resident: zero weight
// traffic inside the 512-step loop.
// ---------------------------------------------------------------------------
__global__ __launch_bounds__(256)
void recurrent_kernel(const float* __restrict__ input,
                      const float* __restrict__ b_ih,
                      const float* __restrict__ b_hh,
                      const __bf16* __restrict__ wsIH,
                      const __bf16* __restrict__ wsHH,
                      const float* __restrict__ attn,
                      float* __restrict__ hx,        // [2][16][16][256] fp32
                      unsigned* __restrict__ bar,    // 16 counters, stride 16
                      float* __restrict__ outW,
                      float* __restrict__ outE) {
  __shared__ __bf16 ihL[16][KT * 512]; // 128KB: W_ih slices, lnt-major
  __shared__ __bf16 hhL[16][KT * 512]; // 128KB: W_hh slices
  __shared__ __bf16 wA[16][WAP];       // weighted, bf16 A operand (padded)
  __shared__ __bf16 hA[16][WAP];       // h, bf16 A operand (padded)
  __shared__ float  gatesL[16][GLP];   // local 256 gate cols, fp32

  const int tid  = threadIdx.x;
  const int lane = tid & 31;
  const int wave = tid >> 5;             // 0..7
  const int row  = tid >> 4;             // 0..15 (elementwise row)
  const int sub  = tid & 15;
  const int g    = blockIdx.x >> 2;      // group: batch tile
  const int s    = blockIdx.x & 3;       // partner: nt mod 4
  const int bgl  = g * 16 + row;

  // ---- stage weight slices into LDS (nt = s + 4*lnt, 8KB contiguous each)
  for (int j = 0; j < 16; ++j) {
    const uint4* sI = (const uint4*)(wsIH + (size_t)(s + 4 * j) * KT * 512);
    const uint4* sH = (const uint4*)(wsHH + (size_t)(s + 4 * j) * KT * 512);
    uint4* dI = (uint4*)&ihL[j][0];
    uint4* dH = (uint4*)&hhL[j][0];
    for (int u = tid; u < 512; u += 256) { dI[u] = sI[u]; dH[u] = sH[u]; }
  }

  // per-thread time-invariant constants
  float attn_reg[16];
  {
    const float* ap = attn + bgl * Nin + sub * 16;
    #pragma unroll
    for (int i = 0; i < 16; ++i) attn_reg[i] = ap[i];
  }
  float bias_reg[2];
  #pragma unroll
  for (int q = 0; q < 2; ++q) {
    const int j = (s + 4 * (2 * wave + q)) * 16 + (lane & 15);
    bias_reg[q] = b_ih[j] + b_hh[j];
  }

  // state: c (4 hidden units / thread) in registers; h = 0
  float c_reg[4];
  #pragma unroll
  for (int a = 0; a < 4; ++a) c_reg[a] = 0.0f;
  {
    __bf16* hp = &hA[row][sub * 16];
    __bf16* wp0 = &wA[row][sub * 16];
    const float* xp0 = input + (size_t)bgl * TM1 * Nin + sub * 16;
    float* op0 = outW + (size_t)bgl * TM1 * Nin + sub * 16;
    #pragma unroll
    for (int i = 0; i < 16; ++i) hp[i] = (__bf16)0.0f;
    #pragma unroll
    for (int i = 0; i < 16; ++i) {        // phase-1 for t = 0
      const float w = attn_reg[i] * xp0[i];
      op0[i] = w;
      wp0[i] = (__bf16)w;
    }
  }
  __syncthreads();

  const float* xrow = input + (size_t)bgl * TM1 * Nin + sub * 16;
  float* owrow = outW + (size_t)bgl * TM1 * Nin + sub * 16;
  float* oerow = outE + (size_t)bgl * TM1 * Hh;       // + t*Hh + j
  unsigned* mybar = bar + g * 16;                     // padded counter

  const int mr  = lane & 15;
  const int klo = (lane < 16) ? 0 : 8;

  for (int t = 0; t < TM1; ++t) {
    // ---- GEMM: local gates = wA @ Wih^T + hA @ Whh^T + bias
    v16bf aWf[KT], aHf[KT];
    #pragma unroll
    for (int kt = 0; kt < KT; ++kt) {
      const int ks1 = kt * 32 + klo;
      const int ks2 = ks1 + 16;
      aWf[kt] = cat16(*(const v8bf*)&wA[mr][ks1], *(const v8bf*)&wA[mr][ks2]);
      aHf[kt] = cat16(*(const v8bf*)&hA[mr][ks1], *(const v8bf*)&hA[mr][ks2]);
    }
    #pragma unroll 1
    for (int q = 0; q < 2; ++q) {
      const int lnt = 2 * wave + q;
      v8f acc;
      #pragma unroll
      for (int r = 0; r < 8; ++r) acc[r] = bias_reg[q];
      #pragma unroll
      for (int kt = 0; kt < KT; ++kt) {
        const __bf16* fbi = &ihL[lnt][kt * 512 + lane * 16];
        const __bf16* fbh = &hhL[lnt][kt * 512 + lane * 16];
        v16bf bI = cat16(*(const v8bf*)fbi, *(const v8bf*)(fbi + 8));
        v16bf bH = cat16(*(const v8bf*)fbh, *(const v8bf*)(fbh + 8));
        acc = __builtin_amdgcn_wmma_f32_16x16x32_bf16(
                  false, aWf[kt], false, bI, (short)0, acc, false, false);
        acc = __builtin_amdgcn_wmma_f32_16x16x32_bf16(
                  false, aHf[kt], false, bH, (short)0, acc, false, false);
      }
      const int lncol = lnt * 16 + (lane & 15);
      const int mrow = (lane < 16) ? 0 : 8;
      #pragma unroll
      for (int r = 0; r < 8; ++r) gatesL[mrow + r][lncol] = acc[r];
    }
    __syncthreads();

    // ---- local LSTM cell update for this WG's 64 hidden units
    {
      float* hxw = hx + ((size_t)((t & 1) * 16 + g) * 16 + row) * 256;
      float* oe = oerow + (size_t)t * Hh;
      float hvv[4];
      #pragma unroll
      for (int a = 0; a < 4; ++a) {
        const float ig = gatesL[row][a * 16 + sub];
        const float fg = gatesL[row][a * 16 + sub + 64];
        const float gg = gatesL[row][a * 16 + sub + 128];
        const float og = gatesL[row][a * 16 + sub + 192];
        const float cs = sigf(fg) * c_reg[a] + sigf(ig) * tanhfast(gg);
        c_reg[a] = cs;
        hvv[a] = sigf(og) * tanhfast(cs);
      }
      #pragma unroll
      for (int a = 0; a < 4; ++a) {
        const int j = (s + 4 * a) * 16 + sub;
        hxw[j] = hvv[a];
        oe[j] = hvv[a];
      }
      // weighted for step t+1 (independent of the exchange)
      if (t + 1 < TM1) {
        const float* xp = xrow + (size_t)(t + 1) * Nin;
        float* op = owrow + (size_t)(t + 1) * Nin;
        __bf16* wp = &wA[row][sub * 16];
        #pragma unroll
        for (int i = 0; i < 16; ++i) {
          const float w = attn_reg[i] * xp[i];
          op[i] = w;
          wp[i] = (__bf16)w;
        }
      }
    }
    __threadfence();
    __syncthreads();

    // ---- 4-WG group barrier (monotonic counter, release/acquire)
    if (tid == 0) {
      __hip_atomic_fetch_add(mybar, 1u, __ATOMIC_RELEASE,
                             __HIP_MEMORY_SCOPE_AGENT);
      const unsigned tgt = 4u * (unsigned)(t + 1);
      while (__hip_atomic_load(mybar, __ATOMIC_ACQUIRE,
                               __HIP_MEMORY_SCOPE_AGENT) < tgt) {
        __builtin_amdgcn_s_sleep(2);
      }
    }
    __syncthreads();

    // ---- rebuild full h (bf16 A operand) from the exchange buffer
    {
      const float* hxr = hx + ((size_t)((t & 1) * 16 + g) * 16 + row) * 256
                            + sub * 16;
      __bf16* hp = &hA[row][sub * 16];
      #pragma unroll
      for (int i = 0; i < 16; ++i) hp[i] = (__bf16)hxr[i];
    }
    __syncthreads();
  }
}

// ---------------------------------------------------------------------------
extern "C" void kernel_launch(void* const* d_in, const int* in_sizes, int n_in,
                              void* d_out, int out_size, void* d_ws, size_t ws_size,
                              hipStream_t stream) {
  (void)in_sizes; (void)n_in; (void)out_size; (void)ws_size;
  const float* input  = (const float*)d_in[0];
  const float* W_ih   = (const float*)d_in[1];
  const float* W_hh   = (const float*)d_in[2];
  const float* b_ih   = (const float*)d_in[3];
  const float* b_hh   = (const float*)d_in[4];
  const float* attn_w = (const float*)d_in[5];
  // d_in[6] (attn_b) unused: softmax is shift-invariant.

  // ws: [wsIH 512KB][wsHH 512KB][attn 256KB][hx 512KB][bar 1KB]
  __bf16* wsIH = (__bf16*)d_ws;
  __bf16* wsHH = wsIH + (size_t)NT * KT * 512;
  float*  attn = (float*)(wsHH + (size_t)NT * KT * 512);
  float*  hx   = attn + (size_t)Bsz * Nin;
  unsigned* bar = (unsigned*)(hx + (size_t)2 * 16 * 16 * 256);

  float* outW = (float*)d_out;
  float* outE = outW + (size_t)Bsz * TM1 * Nin;

  attn_softmax_kernel<<<Bsz, 256, 0, stream>>>(input, attn_w, attn);
  pack_weights_kernel<<<dim3(NT * KT, 2), 512, 0, stream>>>(W_ih, W_hh,
                                                            wsIH, wsHH, bar);
  recurrent_kernel<<<64, 256, 0, stream>>>(input, b_ih, b_hh,
                                           wsIH, wsHH, attn, hx, bar,
                                           outW, outE);
}
